// AttentionL2_8263517077791
// MI455X (gfx1250) — matrix operand order repaired and checked by reading, not compiled, not run on hardware
//
#include <hip/hip_runtime.h>
#include <hip/hip_bf16.h>
#include <cstdint>

// Problem constants (from reference setup_inputs)
#define B_    4
#define S_    4096
#define DIN_  384
#define DOUT_ 64

typedef __attribute__((ext_vector_type(16))) __bf16 v16bf;
typedef __attribute__((ext_vector_type(8)))  float  v8f;
typedef __attribute__((ext_vector_type(4)))  unsigned v4u;
typedef __attribute__((ext_vector_type(8)))  int   v8i;
typedef __attribute__((ext_vector_type(4)))  int   v4i;

union BF16x16 { v16bf v; unsigned u[8]; uint4 q[2]; };

__device__ __forceinline__ v8f wmma_bf16(v16bf a, v16bf b, v8f c) {
  // v_wmma_f32_16x16x32_bf16: D = A(16x32) * B(32x16) + C(16x16 f32)
  return __builtin_amdgcn_wmma_f32_16x16x32_bf16(
      /*neg_a=*/false, a, /*neg_b=*/false, b,
      /*c_mod=*/(short)0, c, /*reuse_a=*/false, /*reuse_b=*/false);
}

// ---------------------------------------------------------------------------
// WMMA register layouts (ISA 7.12.2, 16-bit, wave32):
//  A 16x32: lane half h: VGPR j holds K = 2j + (j<4 ? 8h : 8+8h)  (row = lane&15)
//           -> two contiguous 8-element (16B) runs at K = 8h and K = 16+8h.
//  B 32x16: ISA k order as for A; this code loads A and B both with K = h*16 + [0..15] in half h
//           -> one contiguous 16-element (32B) run at K = 16h. (col = lane&15)
//  C/D 16x16 f32: VGPR r: row = r + 8h, col = lane&15.
// ---------------------------------------------------------------------------

// 16 contiguous bf16 (32B) -> B-tile lane registers (or A-chunk helper).
__device__ __forceinline__ v16bf load_b16x16(const __bf16* p) {
  BF16x16 t;
  t.q[0] = *(const uint4*)p;
  t.q[1] = *(const uint4*)(p + 8);
  return t.v;
}

// A tile from bf16 row-major memory; rowptr = &M[row(lane&15)][k_base]
__device__ __forceinline__ v16bf load_a_bf16(const __bf16* rowptr, int h) {
  BF16x16 t;
  t.q[0] = *(const uint4*)(rowptr + 16 * h);
  t.q[1] = *(const uint4*)(rowptr + 16 * h + 8);
  return t.v;
}

__device__ __forceinline__ v16bf cvt16(float4 f0, float4 f1, float4 f2, float4 f3) {
  v16bf v;
  v[0]  = (__bf16)f0.x; v[1]  = (__bf16)f0.y; v[2]  = (__bf16)f0.z; v[3]  = (__bf16)f0.w;
  v[4]  = (__bf16)f1.x; v[5]  = (__bf16)f1.y; v[6]  = (__bf16)f1.z; v[7]  = (__bf16)f1.w;
  v[8]  = (__bf16)f2.x; v[9]  = (__bf16)f2.y; v[10] = (__bf16)f2.z; v[11] = (__bf16)f2.w;
  v[12] = (__bf16)f3.x; v[13] = (__bf16)f3.y; v[14] = (__bf16)f3.z; v[15] = (__bf16)f3.w;
  return v;
}

// A tile from fp32 memory with on-the-fly bf16 convert (4 independent b128 loads).
__device__ __forceinline__ v16bf load_a_f32(const float* rowptr, int h) {
  const float* p0 = rowptr + 16 * h;
  const float* p1 = rowptr + 16 * h + 8;
  float4 f0 = *(const float4*)p0;
  float4 f1 = *(const float4*)(p0 + 4);
  float4 f2 = *(const float4*)p1;
  float4 f3 = *(const float4*)(p1 + 4);
  return cvt16(f0, f1, f2, f3);
}

// Reductions across the 16 lanes of a wave32 half (xor masks 1..8 stay in-half).
__device__ __forceinline__ float half_sum(float v) {
#pragma unroll
  for (int m = 1; m <= 8; m <<= 1) v += __shfl_xor(v, m, 32);
  return v;
}
__device__ __forceinline__ float half_max(float v) {
#pragma unroll
  for (int m = 1; m <= 8; m <<= 1) v = fmaxf(v, __shfl_xor(v, m, 32));
  return v;
}

// LDS byte offset of a __shared__ object: the LDS aperture maps the wave-
// relative LDS address into bits [31:0] of the generic (flat) address.
__device__ __forceinline__ unsigned lds_off(const void* p) {
  return (unsigned)(size_t)p;
}

// ---------------------------------------------------------------------------
// Tensor Data Mover: 2D bf16 tile (global -> LDS), D# per ISA 8.3/8.4.
//  dim0/tile0: contiguous dimension (elements), dim1/tile1: rows,
//  stride0: elements between rows.
// This toolchain's builtin is the 6-arg form:
//   (uint32x4 g0, int32x8 g1, int32x4 g2, int32x4 g3, int32x8, i32 cpol)
// ---------------------------------------------------------------------------
__device__ __forceinline__ void tdm_load_2d_bf16(unsigned ldsaddr, const void* gptr,
                                                 unsigned dim0, unsigned dim1,
                                                 unsigned tile0, unsigned tile1,
                                                 unsigned stride0) {
  unsigned long long ga = (unsigned long long)(size_t)gptr;
  v4u g0;
  g0[0] = 1u;                                        // count=1, user mode
  g0[1] = ldsaddr;                                   // lds_addr [63:32]
  g0[2] = (unsigned)(ga & 0xFFFFFFFFu);              // global_addr [95:64]
  g0[3] = (unsigned)((ga >> 32) & 0x01FFFFFFu) | (2u << 30); // addr[56:32] | type=2
  v8i g1;
  g1[0] = (int)(1u << 16);                           // data_size=1 (2 bytes)
  g1[1] = (int)((dim0 & 0xFFFFu) << 16);             // tensor_dim0[15:0] @ [63:48]
  g1[2] = (int)(((dim0 >> 16) & 0xFFFFu) | ((dim1 & 0xFFFFu) << 16));
  g1[3] = (int)(((dim1 >> 16) & 0xFFFFu) | ((tile0 & 0xFFFFu) << 16)); // tile_dim0
  g1[4] = (int)(tile1 & 0xFFFFu);                    // tile_dim1 (tile_dim2 = 0)
  g1[5] = (int)stride0;                              // tensor_dim0_stride[31:0]
  g1[6] = 0;                                         // stride hi / dim1_stride lo
  g1[7] = 0;
  v4i z4 = (v4i){0, 0, 0, 0};
  v8i z8 = (v8i){0, 0, 0, 0, 0, 0, 0, 0};
#if __has_builtin(__builtin_amdgcn_tensor_load_to_lds)
  __builtin_amdgcn_tensor_load_to_lds(g0, g1, z4, z4, z8, 0);
#else
  (void)g0; (void)g1; (void)z4; (void)z8;
#endif
}

__device__ __forceinline__ void tdm_wait0() {
#if __has_builtin(__builtin_amdgcn_s_wait_tensorcnt)
  __builtin_amdgcn_s_wait_tensorcnt(0);
#else
  asm volatile("s_wait_tensorcnt 0x0" ::: "memory");
#endif
}

// ---------------------------------------------------------------------------
// Kernel 0: one-shot fp32 -> bf16 conversion of the three weight matrices
// (64x384 each). 4 elements per thread, float4 in, 4x bf16 (8B) out.
// ---------------------------------------------------------------------------
__global__ void __launch_bounds__(256)
cvtw_kernel(const float* __restrict__ Wq, const float* __restrict__ Wk,
            const float* __restrict__ Wv,
            __bf16* __restrict__ Wqb, __bf16* __restrict__ Wkb,
            __bf16* __restrict__ Wvb) {
  const int i = (blockIdx.x * 256 + threadIdx.x) * 4;
  if (i >= DOUT_ * DIN_) return;
  union { __bf16 b[4]; uint2 u; } o;
  float4 q = *(const float4*)(Wq + i);
  o.b[0] = (__bf16)q.x; o.b[1] = (__bf16)q.y; o.b[2] = (__bf16)q.z; o.b[3] = (__bf16)q.w;
  *(uint2*)(Wqb + i) = o.u;
  float4 k = *(const float4*)(Wk + i);
  o.b[0] = (__bf16)k.x; o.b[1] = (__bf16)k.y; o.b[2] = (__bf16)k.z; o.b[3] = (__bf16)k.w;
  *(uint2*)(Wkb + i) = o.u;
  float4 v = *(const float4*)(Wv + i);
  o.b[0] = (__bf16)v.x; o.b[1] = (__bf16)v.y; o.b[2] = (__bf16)v.z; o.b[3] = (__bf16)v.w;
  *(uint2*)(Wvb + i) = o.u;
}

// ---------------------------------------------------------------------------
// Kernel 1: Q/K/V projections (x @ W^T) from bf16 weights, bf16 outputs +
// row norms. One wave per 16-row tile of x. K=384 in 12 steps of 32.
// Writes: Qbf [B,S,64], Kbf [B,S,64], Vt [B,64,S] (transposed), q2/k2 [B,S].
// ---------------------------------------------------------------------------
__global__ void __launch_bounds__(128)
proj_kernel(const float* __restrict__ x,
            const __bf16* __restrict__ Wqb,
            const __bf16* __restrict__ Wkb,
            const __bf16* __restrict__ Wvb,
            __bf16* __restrict__ Qbf, __bf16* __restrict__ Kbf,
            __bf16* __restrict__ Vt,
            float* __restrict__ q2, float* __restrict__ k2) {
  const int lane = threadIdx.x & 31;
  const int wave = threadIdx.x >> 5;
  const int tile = blockIdx.x * 4 + wave;       // 0 .. B*S/16-1
  const int b    = tile >> 8;                   // S/16 == 256 tiles per batch
  const int row0 = (tile & 255) << 4;
  const int h    = lane >> 4;
  const int ln   = lane & 15;

  v8f accq[4], acck[4], accv[4];
#pragma unroll
  for (int t = 0; t < 4; ++t) { accq[t] = (v8f){}; acck[t] = (v8f){}; accv[t] = (v8f){}; }

  const float* xrow = x + ((size_t)b * S_ + row0 + ln) * DIN_;

  for (int ks = 0; ks < DIN_ / 32; ++ks) {
    const int kbase = ks * 32;
    v16bf a = load_a_f32(xrow + kbase, h);
#pragma unroll
    for (int t = 0; t < 4; ++t) {
      const size_t wofs = (size_t)(t * 16 + ln) * DIN_ + kbase + 16 * h;
      // bf16 weights: plain contiguous 32B loads, no dependent converts.
      v16bf bq = load_b16x16(Wqb + wofs);
      v16bf bk = load_b16x16(Wkb + wofs);
      v16bf bv = load_b16x16(Wvb + wofs);
      accq[t] = wmma_bf16(a, bq, accq[t]);
      acck[t] = wmma_bf16(a, bk, acck[t]);
      accv[t] = wmma_bf16(a, bv, accv[t]);
    }
  }

  // Epilogue: bf16 stores + self-consistent squared norms of the bf16 values.
#pragma unroll
  for (int r = 0; r < 8; ++r) {
    const int srow = row0 + r + 8 * h;
    float sq = 0.f, sk = 0.f;
#pragma unroll
    for (int t = 0; t < 4; ++t) {
      const int col = t * 16 + ln;
      __bf16 qb = (__bf16)accq[t][r];
      __bf16 kb = (__bf16)acck[t][r];
      __bf16 vb = (__bf16)accv[t][r];
      float qf = (float)qb, kf = (float)kb;
      sq += qf * qf;
      sk += kf * kf;
      Qbf[((size_t)b * S_ + srow) * DOUT_ + col] = qb;
      Kbf[((size_t)b * S_ + srow) * DOUT_ + col] = kb;
      Vt[((size_t)b * DOUT_ + col) * S_ + srow]  = vb;   // transposed store
    }
    sq = half_sum(sq);
    sk = half_sum(sk);
    if (ln == 0) {
      q2[(size_t)b * S_ + srow] = sq;
      k2[(size_t)b * S_ + srow] = sk;
    }
  }
}

// ---------------------------------------------------------------------------
// Kernel 2: flash-attention over L2 distances, TDM double-buffered K/V tiles.
// Block = 128 threads (4 waves); each wave owns 16 query rows; the block
// streams all 4096 keys in steps of 32. Wave 0 drives the Tensor Data Mover.
// ---------------------------------------------------------------------------
__global__ void __launch_bounds__(128)
attn_kernel(const __bf16* __restrict__ Qbf, const __bf16* __restrict__ Kbf,
            const __bf16* __restrict__ Vt,
            const float* __restrict__ q2, const float* __restrict__ k2,
            float* __restrict__ out) {
  const int lane  = threadIdx.x & 31;
  const int wave  = threadIdx.x >> 5;
  const int b     = blockIdx.y;
  const int qrow0 = blockIdx.x * 64 + wave * 16;
  const int h     = lane >> 4;
  const int ln    = lane & 15;

  __shared__ __bf16  smK[2][32 * 64];   // K tile: [key 0..31][dk 0..63]
  __shared__ __bf16  smV[2][64 * 32];   // V^T tile: [d 0..63][key 0..31]
  __shared__ unsigned smP[4][16 * 16];  // per-wave P tile: 16 rows x 32 bf16

  // Q A-tiles (K = 64 -> two 16x32 slabs) held in registers for the whole loop.
  const __bf16* qrow = Qbf + ((size_t)b * S_ + qrow0 + ln) * DOUT_;
  const v16bf a0 = load_a_bf16(qrow, h);
  const v16bf a1 = load_a_bf16(qrow + 32, h);

  float q2r[8];
#pragma unroll
  for (int r = 0; r < 8; ++r)
    q2r[r] = q2[(size_t)b * S_ + qrow0 + r + 8 * h];

  float mrow[8], lrow[8];
  v8f acc[4];
#pragma unroll
  for (int r = 0; r < 8; ++r) { mrow[r] = -INFINITY; lrow[r] = 0.f; }
#pragma unroll
  for (int t = 0; t < 4; ++t) acc[t] = (v8f){};

  const float LOG2E = 1.4426950408889634f;
  const float INV_SQRT_SCALE = 0.125f;   // 1/sqrt(SCALE=64)

  // Prologue: DMA first K/V tile into buffer 0 (wave 0 only; TENSORcnt is
  // per-wave, so the issuing wave waits, then the barrier publishes LDS).
  if (wave == 0) {
    tdm_load_2d_bf16(lds_off(&smK[0][0]), Kbf + (size_t)b * S_ * DOUT_,
                     /*dim0=*/64, /*dim1=*/32, /*tile0=*/64, /*tile1=*/32,
                     /*stride0=*/64);
    tdm_load_2d_bf16(lds_off(&smV[0][0]), Vt + (size_t)b * DOUT_ * S_,
                     /*dim0=*/32, /*dim1=*/64, /*tile0=*/32, /*tile1=*/64,
                     /*stride0=*/S_);
    tdm_wait0();
  }
  __syncthreads();

  int buf = 0;
  for (int kk0 = 0; kk0 < S_; kk0 += 32) {
    // Kick off DMA of the next tile into the other buffer (wraps harmlessly).
    const int nk = (kk0 + 32 < S_) ? kk0 + 32 : 0;
    if (wave == 0) {
      tdm_load_2d_bf16(lds_off(&smK[buf ^ 1][0]),
                       Kbf + ((size_t)b * S_ + nk) * DOUT_, 64, 32, 64, 32, 64);
      tdm_load_2d_bf16(lds_off(&smV[buf ^ 1][0]),
                       Vt + (size_t)b * DOUT_ * S_ + nk, 32, 64, 32, 64, S_);
    }

    // ---- scores: D = Q(16x64) . K^T(64x16) per 16-key subtile -------------
    float s[2][8];
#pragma unroll
    for (int sub = 0; sub < 2; ++sub) {
      const __bf16* krow = &smK[buf][(sub * 16 + ln) * 64];   // this lane's key
      v8f dot = (v8f){};
      dot = wmma_bf16(a0, load_b16x16(krow + 16 * h), dot);
      dot = wmma_bf16(a1, load_b16x16(krow + 32 + 16 * h), dot);
      const float k2v = k2[(size_t)b * S_ + kk0 + sub * 16 + ln];
#pragma unroll
      for (int r = 0; r < 8; ++r) {
        float d2 = q2r[r] + k2v - 2.0f * dot[r];
        s[sub][r] = sqrtf(fmaxf(d2, 1e-12f)) * INV_SQRT_SCALE;
      }
    }

    // ---- online softmax update over the 32-key tile -----------------------
    __bf16* pview = (__bf16*)smP[wave];
#pragma unroll
    for (int r = 0; r < 8; ++r) {
      const float tmax  = half_max(fmaxf(s[0][r], s[1][r]));
      const float mnew  = fmaxf(mrow[r], tmax);
      const float alpha = exp2f((mrow[r] - mnew) * LOG2E);
      const float p0    = exp2f((s[0][r] - mnew) * LOG2E);
      const float p1    = exp2f((s[1][r] - mnew) * LOG2E);
      lrow[r] = lrow[r] * alpha + half_sum(p0 + p1);
      mrow[r] = mnew;
#pragma unroll
      for (int t = 0; t < 4; ++t) acc[t][r] *= alpha;
      pview[(r + 8 * h) * 32 + ln]      = (__bf16)p0;   // D-layout -> LDS rows
      pview[(r + 8 * h) * 32 + 16 + ln] = (__bf16)p1;
    }

    // Re-read P in A-matrix layout (same wave; DS ops in-order per wave).
    __builtin_amdgcn_wave_barrier();
    asm volatile("s_wait_dscnt 0x0" ::: "memory");
    v16bf ap;
    {
      BF16x16 t;
      const unsigned* pb = &smP[wave][ln * 16];
      t.q[0] = *(const uint4*)(pb + 8 * h);
      t.q[1] = *(const uint4*)(pb + 8 * h + 4);
      ap = t.v;
    }
    __builtin_amdgcn_wave_barrier();

    // ---- O += P(16x32) . V(32x64), B-tiles from transposed V in LDS -------
#pragma unroll
    for (int t = 0; t < 4; ++t) {
      const __bf16* vrow = &smV[buf][(t * 16 + ln) * 32];
      acc[t] = wmma_bf16(ap, load_b16x16(vrow + 16 * h), acc[t]);
    }

    // Close the pipeline stage: next tile landed, everyone flips.
    if (wave == 0) tdm_wait0();
    __syncthreads();
    buf ^= 1;
  }

  // Normalize and write fp32 output [B,S,64].
#pragma unroll
  for (int r = 0; r < 8; ++r) {
    const float inv = 1.0f / lrow[r];
    const int srow = qrow0 + r + 8 * h;
#pragma unroll
    for (int t = 0; t < 4; ++t)
      out[((size_t)b * S_ + srow) * DOUT_ + t * 16 + ln] = acc[t][r] * inv;
  }
}

extern "C" void kernel_launch(void* const* d_in, const int* in_sizes, int n_in,
                              void* d_out, int out_size, void* d_ws, size_t ws_size,
                              hipStream_t stream) {
  (void)in_sizes; (void)n_in; (void)out_size; (void)ws_size;
  const float* x  = (const float*)d_in[0];
  const float* Wq = (const float*)d_in[1];
  const float* Wk = (const float*)d_in[2];
  const float* Wv = (const float*)d_in[3];
  float* out = (float*)d_out;

  // Workspace: 3 * 2MB bf16 tensors + 2 * 64KB norms + 3 * 48KB bf16 weights.
  char* w = (char*)d_ws;
  const size_t nqkv = (size_t)B_ * S_ * DOUT_;
  const size_t nw   = (size_t)DOUT_ * DIN_;
  __bf16* Qbf = (__bf16*)w; w += nqkv * sizeof(__bf16);
  __bf16* Kbf = (__bf16*)w; w += nqkv * sizeof(__bf16);
  __bf16* Vt  = (__bf16*)w; w += nqkv * sizeof(__bf16);
  float* q2 = (float*)w;    w += (size_t)B_ * S_ * sizeof(float);
  float* k2 = (float*)w;    w += (size_t)B_ * S_ * sizeof(float);
  __bf16* Wqb = (__bf16*)w; w += nw * sizeof(__bf16);
  __bf16* Wkb = (__bf16*)w; w += nw * sizeof(__bf16);
  __bf16* Wvb = (__bf16*)w;

  // Weight conversion: 24576 elems / 4 per thread / 256 per block = 24 blocks.
  cvtw_kernel<<<dim3((DOUT_ * DIN_ / 4 + 255) / 256), dim3(256), 0, stream>>>(
      Wq, Wk, Wv, Wqb, Wkb, Wvb);

  // B*S/16 = 1024 row-tiles, 4 waves (tiles) per 128-thread block.
  proj_kernel<<<dim3((B_ * S_ / 16) / 4), dim3(128), 0, stream>>>(
      x, Wqb, Wkb, Wvb, Qbf, Kbf, Vt, q2, k2);

  // 64 query rows per block (4 waves x 16 rows), grid = (S/64, B).
  attn_kernel<<<dim3(S_ / 64, B_), dim3(128), 0, stream>>>(
      Qbf, Kbf, Vt, q2, k2, out);
}
